// LinearCrossAttention_56813827392083
// MI455X (gfx1250) — matrix operand order repaired and checked by reading, not compiled, not run on hardware
//
#include <hip/hip_runtime.h>

#define BB  4
#define T1N 4096
#define T2N 2048
#define CC  512
#define HH  8
#define DDIM 64
#define NINP 3
#define M1  (BB * T1N)   // 16384
#define M2  (BB * T2N)   // 8192

typedef __attribute__((ext_vector_type(16))) _Float16 v16h;
typedef __attribute__((ext_vector_type(8)))  _Float16 v8h;
typedef __attribute__((ext_vector_type(8)))  float    v8f;

// ---- WMMA fragment loaders (16-bit 16x16x32; contiguous k order used for both A and B, not the ISA table order) ----
// A (MxK = 16x32): lane<16 -> row=lane, k=0..15 contiguous; lane>=16 -> row=lane-16, k=16..31.
__device__ __forceinline__ v16h loadA_frag(const _Float16* A, int ld, int row0, int k0, int lane) {
    int r  = lane & 15;
    int ks = (lane & 16) ? 16 : 0;
    const _Float16* p = A + (size_t)(row0 + r) * ld + k0 + ks;
    v8h lo = *(const v8h*)(p);
    v8h hi = *(const v8h*)(p + 8);
    v16h a;
#pragma unroll
    for (int j = 0; j < 8; ++j) { a[j] = lo[j]; a[j + 8] = hi[j]; }
    return a;
}

// B (KxN = 32x16) with B[k][n] = W[n][k] (W row-major [N][K]):
// lane<16 -> col n=lane, k=0..15 contiguous; lane>=16 -> n=lane-16, k=16..31.
__device__ __forceinline__ v16h loadB_frag_rows(const _Float16* W, int ld, int n0, int k0, int lane) {
    int n  = n0 + (lane & 15);
    int ks = (lane & 16) ? 16 : 0;
    return *(const v16h*)(W + (size_t)n * ld + k0 + ks);
}

// ---- f32 -> f16 conversion ----
__global__ void lca_cvt_f16(const float* __restrict__ in, _Float16* __restrict__ out, int n) {
    int idx = (blockIdx.x * blockDim.x + threadIdx.x) * 4;
    if (idx + 3 < n) {
        float4 v = *(const float4*)(in + idx);
        out[idx]     = (_Float16)v.x;
        out[idx + 1] = (_Float16)v.y;
        out[idx + 2] = (_Float16)v.z;
        out[idx + 3] = (_Float16)v.w;
    } else {
        for (int j = idx; j < n; ++j) out[j] = (_Float16)in[j];
    }
}

// ---- Tiled WMMA GEMM: out[M,512] = A[M,512] @ W^T + bias ----
// blockDim 128 = 4 waves; block tile 64x64; each wave computes a 32x32 quad of
// 16x16 WMMA tiles (2 A-frags + 2 B-frags -> 4 WMMAs per K-step: 2x data reuse).
// mode: 0 = f16 store, 1 = softmax over the 64-wide head chunk then f16 store,
//       2 = f32 store.
__global__ __launch_bounds__(128) void lca_gemm_wmma(
    const _Float16* __restrict__ A, const _Float16* __restrict__ W,
    const float* __restrict__ bias,
    _Float16* __restrict__ outh, float* __restrict__ outf, int mode)
{
    __shared__ float smem[64][65];
    int lane = threadIdx.x & 31;
    int w    = threadIdx.x >> 5;          // 0..3
    int wr = w >> 1, wc = w & 1;          // 2x2 wave grid
    int m0 = blockIdx.x * 64 + wr * 32;
    int n0 = blockIdx.y * 64 + wc * 32;

    v8f acc00 = {}, acc01 = {}, acc10 = {}, acc11 = {};
    for (int k0 = 0; k0 < CC; k0 += 32) {
        v16h a0 = loadA_frag(A, CC, m0,      k0, lane);
        v16h a1 = loadA_frag(A, CC, m0 + 16, k0, lane);
        v16h b0 = loadB_frag_rows(W, CC, n0,      k0, lane);
        v16h b1 = loadB_frag_rows(W, CC, n0 + 16, k0, lane);
        acc00 = __builtin_amdgcn_wmma_f32_16x16x32_f16(false, a0, false, b0, (short)0, acc00, false, false);
        acc01 = __builtin_amdgcn_wmma_f32_16x16x32_f16(false, a0, false, b1, (short)0, acc01, false, false);
        acc10 = __builtin_amdgcn_wmma_f32_16x16x32_f16(false, a1, false, b0, (short)0, acc10, false, false);
        acc11 = __builtin_amdgcn_wmma_f32_16x16x32_f16(false, a1, false, b1, (short)0, acc11, false, false);
    }

    int nloc = lane & 15;
    int mofs = (lane & 16) ? 8 : 0;
    float bv0 = bias[n0 + nloc];
    float bv1 = bias[n0 + 16 + nloc];

    if (mode == 1) {
#pragma unroll
        for (int r = 0; r < 8; ++r) {
            int lm0 = wr * 32 + r + mofs;
            int lm1 = lm0 + 16;
            int ln0 = wc * 32 + nloc;
            smem[lm0][ln0]      = acc00[r] + bv0;
            smem[lm0][ln0 + 16] = acc01[r] + bv1;
            smem[lm1][ln0]      = acc10[r] + bv0;
            smem[lm1][ln0 + 16] = acc11[r] + bv1;
        }
        __syncthreads();
        if (threadIdx.x < 64) {
            int lm = threadIdx.x;
            int gm = blockIdx.x * 64 + lm;
            float mx = -3.4e38f;
            for (int c = 0; c < 64; ++c) mx = fmaxf(mx, smem[lm][c]);
            float s = 0.f;
            for (int c = 0; c < 64; ++c) s += __expf(smem[lm][c] - mx);
            float inv = 1.f / s;
            _Float16* op = outh + (size_t)gm * CC + blockIdx.y * 64;
            for (int c = 0; c < 64; ++c) op[c] = (_Float16)(__expf(smem[lm][c] - mx) * inv);
        }
    } else if (mode == 0) {
#pragma unroll
        for (int r = 0; r < 8; ++r) {
            int m = m0 + r + mofs;
            outh[(size_t)m * CC + n0 + nloc]             = (_Float16)(acc00[r] + bv0);
            outh[(size_t)m * CC + n0 + 16 + nloc]        = (_Float16)(acc01[r] + bv1);
            outh[(size_t)(m + 16) * CC + n0 + nloc]      = (_Float16)(acc10[r] + bv0);
            outh[(size_t)(m + 16) * CC + n0 + 16 + nloc] = (_Float16)(acc11[r] + bv1);
        }
    } else {
#pragma unroll
        for (int r = 0; r < 8; ++r) {
            int m = m0 + r + mofs;
            outf[(size_t)m * CC + n0 + nloc]             = acc00[r] + bv0;
            outf[(size_t)m * CC + n0 + 16 + nloc]        = acc01[r] + bv1;
            outf[(size_t)(m + 16) * CC + n0 + nloc]      = acc10[r] + bv0;
            outf[(size_t)(m + 16) * CC + n0 + 16 + nloc] = acc11[r] + bv1;
        }
    }
}

// ---- k_sum over T2: ksum[i][b][h][d] ----
__global__ void lca_ksum(const _Float16* __restrict__ Kall, float* __restrict__ ksum) {
    int idx = blockIdx.x * blockDim.x + threadIdx.x;
    if (idx >= NINP * BB * HH * DDIM) return;
    int d = idx & 63;
    int h = (idx >> 6) & 7;
    int b = (idx >> 9) & 3;
    int i = idx >> 11;
    const _Float16* p = Kall + (((size_t)i * BB + b) * T2N) * CC + h * 64 + d;
    float s = 0.f;
    for (int t = 0; t < T2N; ++t) s += (float)p[(size_t)t * CC];
    ksum[idx] = s;
}

// ---- context[i][b][h] = K_h^T @ V_h  (64x64, K-dim = T2) ----
// One block per (i,b,h); K/V tiles staged in LDS; A frag read transposed.
__global__ __launch_bounds__(256) void lca_context_wmma(
    const _Float16* __restrict__ Kall, const _Float16* __restrict__ Vall,
    _Float16* __restrict__ ctx)
{
    __shared__ _Float16 sK[32 * 64];
    __shared__ _Float16 sV[32 * 64];
    int blk = blockIdx.x;              // i*32 + b*8 + h
    int h = blk & 7;
    int b = (blk >> 3) & 3;
    int i = blk >> 5;
    int lane = threadIdx.x & 31;
    int w    = threadIdx.x >> 5;
    int tm  = w >> 1;                  // tiles 2w, 2w+1 share tm
    int tn0 = (2 * w) & 3;
    const _Float16* Kb = Kall + (((size_t)i * BB + b) * T2N) * CC + h * 64;
    const _Float16* Vb = Vall + (((size_t)i * BB + b) * T2N) * CC + h * 64;
    int lrow = threadIdx.x >> 3;       // 0..31
    int lcol = (threadIdx.x & 7) * 8;  // 0..56

    v8f acc0 = {}, acc1 = {};
    for (int k0 = 0; k0 < T2N; k0 += 32) {
        __syncthreads();
        *(v8h*)(sK + lrow * 64 + lcol) = *(const v8h*)(Kb + (size_t)(k0 + lrow) * CC + lcol);
        *(v8h*)(sV + lrow * 64 + lcol) = *(const v8h*)(Vb + (size_t)(k0 + lrow) * CC + lcol);
        __syncthreads();
        int m  = tm * 16 + (lane & 15);
        int ks = (lane & 16) ? 16 : 0;
        v16h a;
#pragma unroll
        for (int j = 0; j < 8; ++j) {
            a[j]     = sK[(ks + j) * 64 + m];        // A[m][k] = K[k][m]
            a[j + 8] = sK[(ks + 8 + j) * 64 + m];
        }
        int kb = (lane & 16) ? 16 : 0;
        int n  = tn0 * 16 + (lane & 15);
        v16h b0, b1;
#pragma unroll
        for (int j = 0; j < 16; ++j) {
            b0[j] = sV[(kb + j) * 64 + n];
            b1[j] = sV[(kb + j) * 64 + n + 16];
        }
        acc0 = __builtin_amdgcn_wmma_f32_16x16x32_f16(false, a, false, b0, (short)0, acc0, false, false);
        acc1 = __builtin_amdgcn_wmma_f32_16x16x32_f16(false, a, false, b1, (short)0, acc1, false, false);
    }
    _Float16* cb = ctx + (size_t)blk * (DDIM * DDIM);
    int mofs = (lane & 16) ? 8 : 0;
#pragma unroll
    for (int r = 0; r < 8; ++r) {
        int m = tm * 16 + r + mofs;
        int n = tn0 * 16 + (lane & 15);
        cb[m * 64 + n]      = (_Float16)acc0[r];
        cb[m * 64 + n + 16] = (_Float16)acc1[r];
    }
}

// ---- attn = q + sum_i (q @ ctx_i) * Dinv_i ;  one block = 32 rows x 64 cols of head h ----
__global__ __launch_bounds__(256) void lca_combine_wmma(
    const _Float16* __restrict__ Q, const _Float16* __restrict__ ctx,
    const float* __restrict__ ksum, _Float16* __restrict__ attn)
{
    __shared__ float dinv[NINP][32];
    int h    = blockIdx.y;
    int row0 = blockIdx.x * 32;
    int b    = row0 / T1N;
    int tid  = threadIdx.x;
    if (tid < NINP * 32) {
        int i  = tid >> 5;
        int lr = tid & 31;
        const _Float16* qp = Q + (size_t)(row0 + lr) * CC + h * 64;
        const float*    kp = ksum + (((size_t)i * BB + b) * HH + h) * 64;
        float s = 0.f;
        for (int d = 0; d < 64; ++d) s += (float)qp[d] * kp[d];
        dinv[i][lr] = 1.f / (s + 1e-8f);
    }
    __syncthreads();

    int lane = tid & 31;
    int w    = tid >> 5;
    int wr = w >> 2, wc = w & 3;
    int t0 = row0 + wr * 16;
    int n0 = wc * 16;
    int mofs = (lane & 16) ? 8 : 0;

    float accf[8];
#pragma unroll
    for (int r = 0; r < 8; ++r) {
        int t = t0 + r + mofs;
        accf[r] = (float)Q[(size_t)t * CC + h * 64 + n0 + (lane & 15)];
    }
    for (int i = 0; i < NINP; ++i) {
        const _Float16* cb = ctx + (((size_t)i * BB + b) * HH + h) * (DDIM * DDIM);
        v8f c = {};
#pragma unroll
        for (int k0 = 0; k0 < 64; k0 += 32) {
            v16h a = loadA_frag(Q, CC, t0, h * 64 + k0, lane);
            int kb = (lane & 16) ? 16 : 0;
            int n  = n0 + (lane & 15);
            v16h bf;
#pragma unroll
            for (int j = 0; j < 16; ++j) bf[j] = cb[(k0 + kb + j) * 64 + n];  // B[k][n] = ctx[k][n]
            c = __builtin_amdgcn_wmma_f32_16x16x32_f16(false, a, false, bf, (short)0, c, false, false);
        }
#pragma unroll
        for (int r = 0; r < 8; ++r) {
            int lr = wr * 16 + r + mofs;
            accf[r] += c[r] * dinv[i][lr];
        }
    }
#pragma unroll
    for (int r = 0; r < 8; ++r) {
        int t = t0 + r + mofs;
        attn[(size_t)t * CC + h * 64 + n0 + (lane & 15)] = (_Float16)accf[r];
    }
}

extern "C" void kernel_launch(void* const* d_in, const int* in_sizes, int n_in,
                              void* d_out, int out_size, void* d_ws, size_t ws_size,
                              hipStream_t stream) {
    const float* x  = (const float*)d_in[0];
    const float* y0 = (const float*)d_in[1];
    const float* y1 = (const float*)d_in[2];
    const float* y2 = (const float*)d_in[3];
    const float* Wq = (const float*)d_in[4];
    const float* bq = (const float*)d_in[5];
    const float* Wk = (const float*)d_in[6];
    const float* bk = (const float*)d_in[7];
    const float* Wv = (const float*)d_in[8];
    const float* bv = (const float*)d_in[9];
    const float* Wo = (const float*)d_in[10];
    const float* bo = (const float*)d_in[11];
    float* out = (float*)d_out;

    size_t off = 0;
    auto carve = [&](size_t bytes) -> void* {
        void* p = (char*)d_ws + off;
        off += (bytes + 255) & ~(size_t)255;
        return p;
    };
    _Float16* xb   = (_Float16*)carve((size_t)M1 * CC * 2);
    _Float16* yb   = (_Float16*)carve((size_t)NINP * M2 * CC * 2);
    _Float16* Wqb  = (_Float16*)carve((size_t)CC * CC * 2);
    _Float16* Wkb  = (_Float16*)carve((size_t)NINP * CC * CC * 2);
    _Float16* Wvb  = (_Float16*)carve((size_t)NINP * CC * CC * 2);
    _Float16* Wob  = (_Float16*)carve((size_t)CC * CC * 2);
    _Float16* Qb   = (_Float16*)carve((size_t)M1 * CC * 2);
    _Float16* Kb   = (_Float16*)carve((size_t)NINP * M2 * CC * 2);
    _Float16* Vb   = (_Float16*)carve((size_t)NINP * M2 * CC * 2);
    float*    ksumf= (float*)   carve((size_t)NINP * BB * HH * DDIM * 4);
    _Float16* ctxb = (_Float16*)carve((size_t)NINP * BB * HH * DDIM * DDIM * 2);
    _Float16* attnb= (_Float16*)carve((size_t)M1 * CC * 2);
    (void)ws_size; (void)in_sizes; (void)n_in; (void)out_size;

    // ---- convert inputs/weights to f16 ----
    {
        int n = M1 * CC;                               // x
        lca_cvt_f16<<<n / 1024, 256, 0, stream>>>(x, xb, n);
    }
    {
        int n = M2 * CC;                               // y0..y2
        lca_cvt_f16<<<n / 1024, 256, 0, stream>>>(y0, yb,                   n);
        lca_cvt_f16<<<n / 1024, 256, 0, stream>>>(y1, yb + (size_t)M2*CC,   n);
        lca_cvt_f16<<<n / 1024, 256, 0, stream>>>(y2, yb + (size_t)2*M2*CC, n);
    }
    {
        int n1 = CC * CC, n3 = NINP * CC * CC;
        lca_cvt_f16<<<n1 / 1024, 256, 0, stream>>>(Wq, Wqb, n1);
        lca_cvt_f16<<<n3 / 1024, 256, 0, stream>>>(Wk, Wkb, n3);
        lca_cvt_f16<<<n3 / 1024, 256, 0, stream>>>(Wv, Wvb, n3);
        lca_cvt_f16<<<n1 / 1024, 256, 0, stream>>>(Wo, Wob, n1);
    }

    // ---- Q = softmax(x @ Wq^T + bq) per head chunk ----
    lca_gemm_wmma<<<dim3(M1 / 64, CC / 64), 128, 0, stream>>>(xb, Wqb, bq, Qb, nullptr, 1);

    // ---- K_i = softmax(y_i @ Wk_i^T + bk_i), V_i = y_i @ Wv_i^T + bv_i ----
    for (int i = 0; i < NINP; ++i) {
        const _Float16* yi = yb + (size_t)i * M2 * CC;
        lca_gemm_wmma<<<dim3(M2 / 64, CC / 64), 128, 0, stream>>>(
            yi, Wkb + (size_t)i * CC * CC, bk + (size_t)i * CC,
            Kb + (size_t)i * M2 * CC, nullptr, 1);
        lca_gemm_wmma<<<dim3(M2 / 64, CC / 64), 128, 0, stream>>>(
            yi, Wvb + (size_t)i * CC * CC, bv + (size_t)i * CC,
            Vb + (size_t)i * M2 * CC, nullptr, 0);
    }

    // ---- k_sum, context, combine ----
    lca_ksum<<<(NINP * BB * HH * DDIM) / 256, 256, 0, stream>>>(Kb, ksumf);
    lca_context_wmma<<<NINP * BB * HH, 256, 0, stream>>>(Kb, Vb, ctxb);
    lca_combine_wmma<<<dim3(M1 / 32, HH), 256, 0, stream>>>(Qb, ctxb, ksumf, attnb);

    // ---- out = attn @ Wo^T + bo (f32) ----
    lca_gemm_wmma<<<dim3(M1 / 64, CC / 64), 128, 0, stream>>>(attnb, Wob, bo, nullptr, out, 2);
}
